// StartPointAttnModelDeep_90074054131990
// MI455X (gfx1250) — compile-verified, matrix operand order checked by reading, not run on hardware
//
#include <hip/hip_runtime.h>
#include <hip/hip_fp16.h>

// ---------------------------------------------------------------------------
// CDNA5 (gfx1250) implementation of the LSTM encoder / attention-decoder.
//   * All GEMMs: v_wmma_f32_16x16x32_f16 (f16 A/B, f32 accumulate).
//   * Both A and B operands of the big time-parallel GEMMs are pre-packed in
//     WMMA fragment layout (f16), so the GEMM inner loop is pure
//     v16h-load + wmma with 2x2 register blocking (4 wmma / 4 frag loads).
//   * Recurrent parts: one persistent workgroup, h in LDS fragments, c in
//     registers; the h sequence is emitted directly as packed A fragments
//     for the next GEMM.
//   * Decoder: 64 sequential (attention, LSTM-step) launch pairs; K/V (48MB)
//     stay resident in the 192MB L2.
// ---------------------------------------------------------------------------

typedef __attribute__((ext_vector_type(16))) _Float16 v16h;
typedef __attribute__((ext_vector_type(8)))  float    v8f;

#define WMMA_F16(a, b, c) \
    __builtin_amdgcn_wmma_f32_16x16x32_f16(false, (a), false, (b), (short)0, (c), false, false)

static constexpr int T_SEQ = 512;   // encoder time steps
static constexpr int BATCH = 64;
static constexpr int H     = 128;
static constexpr int G4H   = 512;   // 4*H (gates)
static constexpr int DSTEP = 64;    // decoder steps
static constexpr int VOCAB = 3;

__device__ inline float sigmoidf_(float x) { return 1.0f / (1.0f + __expf(-x)); }

// Scatter one f32 value into a fragment buffer (f16), A/B fragment layout
// (16-bit operand, 16x32 tile, ISA 7.12.2):
//   lane<16 : row=lane,    halfs = K {0..7, 16..23}
//   lane>=16: row=lane-16, halfs = K {8..15, 24..31}
__device__ inline void scatterFrag(_Float16* frag, int kTiles, int row, int kk, float v) {
    const int m   = row >> 4;
    const int rl  = row & 15;
    const int ks  = kk >> 5;
    const int kof = kk & 31;
    const int grp = (kof >> 3) & 1;                 // which half-lane group holds this K
    const int wl  = rl + (grp ? 16 : 0);            // reader lane
    const int hh  = grp ? ((kof < 16) ? kof - 8 : kof - 16)
                        : ((kof < 8)  ? kof      : kof - 8);
    frag[((size_t)(m * kTiles + ks) * 512) + wl * 16 + hh] = (_Float16)v;
}

// ---------------------------------------------------------------------------
// Pack a row-major f32 matrix (Rows x Kin) into WMMA fragment order, f16.
// Identical index math serves both A fragments (Rows = M) and B fragments
// (Rows = N, i.e. weight W used as W^T).
// Per 16x16x32 tile (rb, kb): 512 halfs ordered [lane][half].
// ---------------------------------------------------------------------------
__global__ void pack_frag(const float* __restrict__ W, _Float16* __restrict__ P,
                          int Rows, int Kin) {
    const int kTiles = Kin >> 5;
    const int total  = Rows * Kin;
    for (int idx = blockIdx.x * blockDim.x + threadIdx.x; idx < total;
         idx += gridDim.x * blockDim.x) {
        const int tile = idx >> 9;
        const int e    = idx & 511;
        const int lane = e >> 4;
        const int h    = e & 15;
        const int rb   = tile / kTiles;
        const int kb   = tile % kTiles;
        int r, k;
        if (lane < 16) { r = rb * 16 + lane;      k = kb * 32 + ((h < 8) ? h : h + 8);  }
        else           { r = rb * 16 + lane - 16; k = kb * 32 + ((h < 8) ? h + 8 : h + 16); }
        P[idx] = (_Float16)W[(size_t)r * Kin + k];
    }
}

// ---------------------------------------------------------------------------
// WMMA GEMM, both operands pre-packed:  C (MxN, f32) = Ap * Bp + bias[n]
// 2x2 tile block (32x32 output) per wave: per k-step 4 v16h loads -> 4 wmma.
// 8 waves / 256-thread block; consecutive waves share the B column pair.
// ---------------------------------------------------------------------------
__global__ void __launch_bounds__(256)
gemm_wmma_packed(const _Float16* __restrict__ Ap, const _Float16* __restrict__ Bp,
                 const float* __restrict__ bias, float* __restrict__ C,
                 int M, int N, int K) {
    const int kTiles = K >> 5;
    const int mT2    = M >> 5;   // 32-row block count
    const int nT2    = N >> 5;   // 32-col block count
    const int lane   = threadIdx.x & 31;
    const int wave   = threadIdx.x >> 5;
    const int tile   = blockIdx.x * 8 + wave;
    if (tile >= mT2 * nT2) return;
    const int tm = (tile % mT2) * 2;
    const int tn = (tile / mT2) * 2;

    const _Float16* a0p = Ap + (size_t)(tm + 0) * kTiles * 512 + lane * 16;
    const _Float16* a1p = Ap + (size_t)(tm + 1) * kTiles * 512 + lane * 16;
    const _Float16* b0p = Bp + (size_t)(tn + 0) * kTiles * 512 + lane * 16;
    const _Float16* b1p = Bp + (size_t)(tn + 1) * kTiles * 512 + lane * 16;

    v8f acc[2][2];
    v8f zero = {};
    acc[0][0] = zero; acc[0][1] = zero; acc[1][0] = zero; acc[1][1] = zero;

    for (int kb = 0; kb < kTiles; ++kb) {
        // prefetch two k-steps ahead (global_prefetch_b8)
        __builtin_prefetch(a0p + 1024, 0, 0);
        __builtin_prefetch(a1p + 1024, 0, 0);
        __builtin_prefetch(b0p + 1024, 0, 0);
        __builtin_prefetch(b1p + 1024, 0, 0);
        const v16h a0 = *(const v16h*)a0p; a0p += 512;
        const v16h a1 = *(const v16h*)a1p; a1p += 512;
        const v16h b0 = *(const v16h*)b0p; b0p += 512;
        const v16h b1 = *(const v16h*)b1p; b1p += 512;
        acc[0][0] = WMMA_F16(a0, b0, acc[0][0]);
        acc[0][1] = WMMA_F16(a0, b1, acc[0][1]);
        acc[1][0] = WMMA_F16(a1, b0, acc[1][0]);
        acc[1][1] = WMMA_F16(a1, b1, acc[1][1]);
    }

    const int ln     = lane & 15;
    const int hiHalf = (lane >= 16);
#pragma unroll
    for (int ti = 0; ti < 2; ++ti) {
#pragma unroll
        for (int tj = 0; tj < 2; ++tj) {
            const int   n  = (tn + tj) * 16 + ln;
            const float bv = bias ? bias[n] : 0.0f;
            const int   r0 = (tm + ti) * 16 + (hiHalf ? 8 : 0);
#pragma unroll
            for (int e = 0; e < 8; ++e)
                C[(size_t)(r0 + e) * N + n] = acc[ti][tj][e] + bv;
        }
    }
}

// ---------------------------------------------------------------------------
// One LSTM layer over T steps.  Single persistent workgroup, 512 threads
// (16 waves).  Xpre already contains x@Wih^T + b.  h kept in LDS A-fragment
// layout; c kept in registers.  Each wave owns 2 groups (m-tile, j-tile),
// each group covering the same 16x16 (batch x hidden) block of all 4 gates.
// The hidden-state sequence is written directly as packed A fragments
// (M = T*64, K = 128 -> kTiles = 4) for the next GEMM.
// ---------------------------------------------------------------------------
__global__ void __launch_bounds__(512)
lstm_layer(const float* __restrict__ Xpre,       // (T, 64, 512)
           const _Float16* __restrict__ Whh_p,   // packed: N=512 (32 nT), K=128 (4 kT)
           _Float16* __restrict__ HoutFrag,      // packed A frags (T*64 x 128)
           int T) {
    __shared__ __align__(32) _Float16 hfrag[4 * 4 * 512];   // 16 KB

    const int tid    = threadIdx.x;
    const int lane   = tid & 31;
    const int wave   = tid >> 5;
    const int ln     = lane & 15;
    const int hiHalf = (lane >= 16);
    const int g0     = wave * 2;

    for (int i = tid; i < 4 * 4 * 512; i += 512) hfrag[i] = (_Float16)0.0f;

    float creg[2][8];
#pragma unroll
    for (int g = 0; g < 2; ++g)
#pragma unroll
        for (int e = 0; e < 8; ++e) creg[g][e] = 0.0f;

    __syncthreads();

    for (int t = 0; t < T; ++t) {
        v8f acc[2][4];
        v8f zero = {};
#pragma unroll
        for (int g = 0; g < 2; ++g)
#pragma unroll
            for (int gi = 0; gi < 4; ++gi) acc[g][gi] = zero;

#pragma unroll
        for (int g = 0; g < 2; ++g) {
            const int gid = g0 + g;
            const int m   = gid >> 3;
            const int j   = gid & 7;
#pragma unroll
            for (int kb = 0; kb < 4; ++kb) {
                const v16h a = *(const v16h*)&hfrag[((m * 4 + kb) * 512) + lane * 16];
#pragma unroll
                for (int gi = 0; gi < 4; ++gi) {
                    const v16h b = *(const v16h*)(Whh_p +
                        (size_t)(((gi * 8 + j) * 4 + kb) * 512 + lane * 16));
                    acc[g][gi] = WMMA_F16(a, b, acc[g][gi]);
                }
            }
        }
        __syncthreads();   // all hfrag reads done before overwrite

#pragma unroll
        for (int g = 0; g < 2; ++g) {
            const int gid = g0 + g;
            const int m   = gid >> 3;
            const int j   = gid & 7;
            const int cj  = j * 16 + ln;
#pragma unroll
            for (int e = 0; e < 8; ++e) {
                const int row = m * 16 + e + (hiHalf ? 8 : 0);
                const float* xp = Xpre + ((size_t)t * BATCH + row) * G4H;
                const float gi_ = sigmoidf_(acc[g][0][e] + xp[cj]);
                const float gf  = sigmoidf_(acc[g][1][e] + xp[128 + cj]);
                const float gg  = tanhf    (acc[g][2][e] + xp[256 + cj]);
                const float go  = sigmoidf_(acc[g][3][e] + xp[384 + cj]);
                const float c   = gf * creg[g][e] + gi_ * gg;
                creg[g][e]      = c;
                const float hv  = go * tanhf(c);
                scatterFrag(hfrag, 4, row, cj, hv);                   // LDS (recurrence)
                scatterFrag(HoutFrag, 4, t * BATCH + row, cj, hv);    // global (next GEMM)
            }
        }
        __syncthreads();   // hfrag writes visible before next step's reads
    }
}

// ---------------------------------------------------------------------------
// Decoder attention step.  One workgroup per batch element (grid = 64).
// K/V are the precomputed in-projection of the encoding: QKV (T*64, 384),
// K at column 128, V at column 256.  K/V total 48MB -> L2-resident.
// ---------------------------------------------------------------------------
__global__ void __launch_bounds__(256)
dec_attention(const float* __restrict__ out_prev,   // (64,128)
              const float* __restrict__ QKV,        // (T*64, 384)
              const float* __restrict__ attn_in_w,  // (384,128); rows 0..127 = Wq
              const float* __restrict__ attn_in_b,  // (384,)
              const float* __restrict__ attn_out_w, // (128,128)
              const float* __restrict__ attn_out_b, // (128,)
              float* __restrict__ xcat,             // (64,256) = [out, ctx]
              int T) {
    __shared__ float s_out[128];
    __shared__ float s_q[128];
    __shared__ float s_scores[4 * 512];
    __shared__ float s_ctx[128];
    __shared__ float s_red[256];
    __shared__ float s_sum[4];

    const int b   = blockIdx.x;
    const int tid = threadIdx.x;

    if (tid < 128) s_out[tid] = out_prev[b * 128 + tid];
    __syncthreads();

    if (tid < 128) {
        float s = attn_in_b[tid];
#pragma unroll 4
        for (int k = 0; k < 128; ++k) s += s_out[k] * attn_in_w[tid * 128 + k];
        s_q[tid] = s * 0.1767766952966369f;    // 1/sqrt(32)
    }
    __syncthreads();

    // scores[b,h,t] = q[h,:] . K[t,b,h,:]
    for (int i = tid; i < 4 * 512; i += 256) {
        const int h = i >> 9;
        const int t = i & 511;
        const float* kp = QKV + ((size_t)t * BATCH + b) * 384 + 128 + h * 32;
        const float* qp = s_q + h * 32;
        float s = 0.0f;
#pragma unroll
        for (int d = 0; d < 32; ++d) s += qp[d] * kp[d];
        s_scores[h * 512 + t] = s;
    }
    __syncthreads();

    // softmax over t (per head), 64 threads per head
    {
        const int h = tid >> 6;
        const int l = tid & 63;
        float m = -3.4e38f;
        for (int t = l; t < 512; t += 64) m = fmaxf(m, s_scores[h * 512 + t]);
        s_red[tid] = m;
        __syncthreads();
        for (int s2 = 32; s2 > 0; s2 >>= 1) {
            if (l < s2) s_red[tid] = fmaxf(s_red[tid], s_red[tid + s2]);
            __syncthreads();
        }
        const float mh = s_red[tid & ~63];
        __syncthreads();
        float sum = 0.0f;
        for (int t = l; t < 512; t += 64) {
            const float e = __expf(s_scores[h * 512 + t] - mh);
            s_scores[h * 512 + t] = e;
            sum += e;
        }
        s_red[tid] = sum;
        __syncthreads();
        for (int s2 = 32; s2 > 0; s2 >>= 1) {
            if (l < s2) s_red[tid] += s_red[tid + s2];
            __syncthreads();
        }
        if (l == 0) s_sum[h] = s_red[tid];
    }
    __syncthreads();

    // ctx[h,d] = sum_t attn * V ; 2 threads per output
    {
        const int o    = tid >> 1;
        const int half = tid & 1;
        const int hh   = o >> 5;
        const int d    = o & 31;
        float acc = 0.0f;
        for (int t = half * 256; t < half * 256 + 256; ++t)
            acc += s_scores[hh * 512 + t] *
                   QKV[((size_t)t * BATCH + b) * 384 + 256 + hh * 32 + d];
        s_red[tid] = acc;
        __syncthreads();
        if (half == 0) s_ctx[o] = (s_red[tid] + s_red[tid + 1]) / s_sum[hh];
    }
    __syncthreads();

    if (tid < 128) {
        float s = attn_out_b[tid];
#pragma unroll 4
        for (int k = 0; k < 128; ++k) s += attn_out_w[tid * 128 + k] * s_ctx[k];
        xcat[b * 256 + 128 + tid] = s;
        xcat[b * 256 + tid]       = s_out[tid];
    }
}

// ---------------------------------------------------------------------------
// One decoder 2-layer LSTM step + output projection.  Single workgroup,
// 512 threads.  Same group scheme as the encoder; both gate GEMMs run on
// WMMA with LDS-resident A fragments.  State lives in workspace (f32).
// ---------------------------------------------------------------------------
__global__ void __launch_bounds__(512)
dec_lstm_step(const float* __restrict__ xcat,      // (64,256)
              const _Float16* __restrict__ Wih0p,  // K=256 (8 kT), N=512
              const _Float16* __restrict__ Whh0p,  // K=128 (4 kT)
              const _Float16* __restrict__ Wih1p,  // K=128 (4 kT)
              const _Float16* __restrict__ Whh1p,  // K=128 (4 kT)
              const float* __restrict__ b0, const float* __restrict__ b1,
              const float* __restrict__ lin_w,     // (3,128)
              const float* __restrict__ lin_b,     // (3,)
              float* __restrict__ h0, float* __restrict__ c0,
              float* __restrict__ h1, float* __restrict__ c1,
              float* __restrict__ out_prev,        // (64,128) = relu(nh1)
              float* __restrict__ y)               // (64,3)
{
    __shared__ __align__(32) _Float16 frags[32768];          // 64 KB
    _Float16* xfrag   = frags;                               // 16384 halfs (kT=8)
    _Float16* h0frag  = frags + 16384;                       // 8192  halfs (kT=4)
    _Float16* h1frag  = frags + 24576;                       // 8192  halfs (kT=4)
    _Float16* nh0frag = frags;                               // reuse xfrag region

    const int tid    = threadIdx.x;
    const int lane   = tid & 31;
    const int wave   = tid >> 5;
    const int ln     = lane & 15;
    const int hiHalf = (lane >= 16);
    const int g0     = wave * 2;

    // stage in A fragments
    for (int idx = tid; idx < BATCH * 256; idx += 512)
        scatterFrag(xfrag, 8, idx >> 8, idx & 255, xcat[idx]);
    for (int idx = tid; idx < BATCH * H; idx += 512)
        scatterFrag(h0frag, 4, idx >> 7, idx & 127, h0[idx]);
    for (int idx = tid; idx < BATCH * H; idx += 512)
        scatterFrag(h1frag, 4, idx >> 7, idx & 127, h1[idx]);
    __syncthreads();

    v8f zero = {};

    // ---- layer 0: gates = xcat@Wih0^T + h0@Whh0^T + b0 ----
    v8f acc[2][4];
#pragma unroll
    for (int g = 0; g < 2; ++g)
#pragma unroll
        for (int gi = 0; gi < 4; ++gi) acc[g][gi] = zero;

#pragma unroll
    for (int g = 0; g < 2; ++g) {
        const int gid = g0 + g;
        const int m = gid >> 3, j = gid & 7;
#pragma unroll
        for (int kb = 0; kb < 8; ++kb) {
            const v16h a = *(const v16h*)&xfrag[((m * 8 + kb) * 512) + lane * 16];
#pragma unroll
            for (int gi = 0; gi < 4; ++gi) {
                const v16h bm = *(const v16h*)(Wih0p +
                    (size_t)(((gi * 8 + j) * 8 + kb) * 512 + lane * 16));
                acc[g][gi] = WMMA_F16(a, bm, acc[g][gi]);
            }
        }
#pragma unroll
        for (int kb = 0; kb < 4; ++kb) {
            const v16h a = *(const v16h*)&h0frag[((m * 4 + kb) * 512) + lane * 16];
#pragma unroll
            for (int gi = 0; gi < 4; ++gi) {
                const v16h bm = *(const v16h*)(Whh0p +
                    (size_t)(((gi * 8 + j) * 4 + kb) * 512 + lane * 16));
                acc[g][gi] = WMMA_F16(a, bm, acc[g][gi]);
            }
        }
    }
    __syncthreads();   // xfrag reads done before nh0frag overwrite

#pragma unroll
    for (int g = 0; g < 2; ++g) {
        const int gid = g0 + g;
        const int m = gid >> 3, j = gid & 7;
        const int cj = j * 16 + ln;
#pragma unroll
        for (int e = 0; e < 8; ++e) {
            const int row = m * 16 + e + (hiHalf ? 8 : 0);
            const float gi_ = sigmoidf_(acc[g][0][e] + b0[cj]);
            const float gf  = sigmoidf_(acc[g][1][e] + b0[128 + cj]);
            const float gg  = tanhf    (acc[g][2][e] + b0[256 + cj]);
            const float go  = sigmoidf_(acc[g][3][e] + b0[384 + cj]);
            const float c   = gf * c0[row * H + cj] + gi_ * gg;
            c0[row * H + cj] = c;
            const float hv = go * tanhf(c);
            h0[row * H + cj] = hv;
            scatterFrag(nh0frag, 4, row, cj, hv);
        }
    }
    __syncthreads();

    // ---- layer 1: gates = nh0@Wih1^T + h1@Whh1^T + b1 ----
#pragma unroll
    for (int g = 0; g < 2; ++g)
#pragma unroll
        for (int gi = 0; gi < 4; ++gi) acc[g][gi] = zero;

#pragma unroll
    for (int g = 0; g < 2; ++g) {
        const int gid = g0 + g;
        const int m = gid >> 3, j = gid & 7;
#pragma unroll
        for (int kb = 0; kb < 4; ++kb) {
            const v16h a0 = *(const v16h*)&nh0frag[((m * 4 + kb) * 512) + lane * 16];
            const v16h a1 = *(const v16h*)&h1frag [((m * 4 + kb) * 512) + lane * 16];
#pragma unroll
            for (int gi = 0; gi < 4; ++gi) {
                const v16h bm0 = *(const v16h*)(Wih1p +
                    (size_t)(((gi * 8 + j) * 4 + kb) * 512 + lane * 16));
                const v16h bm1 = *(const v16h*)(Whh1p +
                    (size_t)(((gi * 8 + j) * 4 + kb) * 512 + lane * 16));
                acc[g][gi] = WMMA_F16(a0, bm0, acc[g][gi]);
                acc[g][gi] = WMMA_F16(a1, bm1, acc[g][gi]);
            }
        }
    }

#pragma unroll
    for (int g = 0; g < 2; ++g) {
        const int gid = g0 + g;
        const int m = gid >> 3, j = gid & 7;
        const int cj = j * 16 + ln;
#pragma unroll
        for (int e = 0; e < 8; ++e) {
            const int row = m * 16 + e + (hiHalf ? 8 : 0);
            const float gi_ = sigmoidf_(acc[g][0][e] + b1[cj]);
            const float gf  = sigmoidf_(acc[g][1][e] + b1[128 + cj]);
            const float gg  = tanhf    (acc[g][2][e] + b1[256 + cj]);
            const float go  = sigmoidf_(acc[g][3][e] + b1[384 + cj]);
            const float c   = gf * c1[row * H + cj] + gi_ * gg;
            c1[row * H + cj] = c;
            const float hv = go * tanhf(c);
            h1[row * H + cj] = hv;                     // hidden state (pre-relu)
            out_prev[row * H + cj] = fmaxf(hv, 0.0f);  // new_out = relu(nh1)
        }
    }
    __syncthreads();   // out_prev visible to whole block

    // y = relu(nh1) @ lin_w^T + lin_b
    if (tid < BATCH * VOCAB) {
        const int bb = tid / VOCAB;
        const int v  = tid - bb * VOCAB;
        float s = lin_b[v];
        const float* op = out_prev + bb * H;
        const float* lw = lin_w + v * H;
#pragma unroll 4
        for (int k = 0; k < H; ++k) s += op[k] * lw[k];
        y[tid] = s;
    }
}

__global__ void zero_f32(float* p, int n) {
    const int i = blockIdx.x * blockDim.x + threadIdx.x;
    if (i < n) p[i] = 0.0f;
}

// ---------------------------------------------------------------------------
// Host launcher
// ---------------------------------------------------------------------------
extern "C" void kernel_launch(void* const* d_in, const int* in_sizes, int n_in,
                              void* d_out, int out_size, void* d_ws, size_t ws_size,
                              hipStream_t stream) {
    (void)in_sizes; (void)n_in; (void)out_size; (void)ws_size;

    const float* cnn        = (const float*)d_in[0];
    const float* enc_Wih0   = (const float*)d_in[1];
    const float* enc_Whh0   = (const float*)d_in[2];
    const float* enc_b0     = (const float*)d_in[3];
    const float* enc_Wih1   = (const float*)d_in[4];
    const float* enc_Whh1   = (const float*)d_in[5];
    const float* enc_b1     = (const float*)d_in[6];
    const float* attn_in_w  = (const float*)d_in[7];
    const float* attn_in_b  = (const float*)d_in[8];
    const float* attn_out_w = (const float*)d_in[9];
    const float* attn_out_b = (const float*)d_in[10];
    const float* dec_Wih0   = (const float*)d_in[11];
    const float* dec_Whh0   = (const float*)d_in[12];
    const float* dec_b0     = (const float*)d_in[13];
    const float* dec_Wih1   = (const float*)d_in[14];
    const float* dec_Whh1   = (const float*)d_in[15];
    const float* dec_b1     = (const float*)d_in[16];
    const float* lin_w      = (const float*)d_in[17];
    const float* lin_b      = (const float*)d_in[18];
    float*       out        = (float*)d_out;

    char* ws = (char*)d_ws;
    size_t off = 0;
    auto take = [&](size_t bytes) -> char* {
        char* r = ws + off;
        off += (bytes + 255) & ~(size_t)255;
        return r;
    };

    const size_t MB_ROWS = (size_t)T_SEQ * BATCH;   // 32768

    // packed weights (B fragments, f16)
    _Float16* pWih0    = (_Float16*)take((size_t)G4H * 1024 * 2);
    _Float16* pWhh0    = (_Float16*)take((size_t)G4H * H * 2);
    _Float16* pWih1    = (_Float16*)take((size_t)G4H * H * 2);
    _Float16* pWhh1    = (_Float16*)take((size_t)G4H * H * 2);
    _Float16* pAttnIn  = (_Float16*)take((size_t)384 * H * 2);
    _Float16* pDecWih0 = (_Float16*)take((size_t)G4H * 256 * 2);
    _Float16* pDecWhh0 = (_Float16*)take((size_t)G4H * H * 2);
    _Float16* pDecWih1 = (_Float16*)take((size_t)G4H * H * 2);
    _Float16* pDecWhh1 = (_Float16*)take((size_t)G4H * H * 2);

    // big shared region: packed cnn A fragments (64MB) is dead after the
    // first GEMM; QKV (48MB) reuses the same space.
    char*     bigRegion = take(MB_ROWS * 1024 * 2);      // 64 MB
    _Float16* pCnnA     = (_Float16*)bigRegion;
    float*    QKV       = (float*)bigRegion;

    float*    Xpre   = (float*)take(MB_ROWS * G4H * 4);  // 64 MB (reused L0/L1)
    _Float16* Hfrag0 = (_Float16*)take(MB_ROWS * H * 2); // 8 MB (packed A frags)
    _Float16* Hfrag1 = (_Float16*)take(MB_ROWS * H * 2); // 8 MB

    float* state = (float*)take((size_t)5 * BATCH * H * 4);
    float* h0 = state;
    float* c0 = state + 1 * BATCH * H;
    float* h1 = state + 2 * BATCH * H;
    float* c1 = state + 3 * BATCH * H;
    float* out_prev = state + 4 * BATCH * H;
    float* xcat = (float*)take((size_t)BATCH * 256 * 4);

    auto packN = [&](const float* W, _Float16* P, int Rows, int Kin) {
        const int total  = Rows * Kin;
        int blocks = (total + 255) / 256;
        if (blocks > 4096) blocks = 4096;
        pack_frag<<<blocks, 256, 0, stream>>>(W, P, Rows, Kin);
    };

    // weight packing (f32 -> f16 fragment layout)
    packN(enc_Wih0,  pWih0,    G4H, 1024);
    packN(enc_Whh0,  pWhh0,    G4H, H);
    packN(enc_Wih1,  pWih1,    G4H, H);
    packN(enc_Whh1,  pWhh1,    G4H, H);
    packN(attn_in_w, pAttnIn,  384, H);
    packN(dec_Wih0,  pDecWih0, G4H, 256);
    packN(dec_Whh0,  pDecWhh0, G4H, H);
    packN(dec_Wih1,  pDecWih1, G4H, H);
    packN(dec_Whh1,  pDecWhh1, G4H, H);
    // cnn_output -> A fragments (same layout math)
    packN(cnn, pCnnA, (int)MB_ROWS, 1024);

    auto gemm = [&](const _Float16* Ap, const _Float16* Bp, const float* bias,
                    float* C, int M, int N, int K) {
        const int tiles  = (M >> 5) * (N >> 5);
        const int blocks = (tiles + 7) / 8;
        gemm_wmma_packed<<<blocks, 256, 0, stream>>>(Ap, Bp, bias, C, M, N, K);
    };

    // ---- encoder ----
    gemm(pCnnA, pWih0, enc_b0, Xpre, (int)MB_ROWS, G4H, 1024);
    lstm_layer<<<1, 512, 0, stream>>>(Xpre, pWhh0, Hfrag0, T_SEQ);
    gemm(Hfrag0, pWih1, enc_b1, Xpre, (int)MB_ROWS, G4H, H);
    lstm_layer<<<1, 512, 0, stream>>>(Xpre, pWhh1, Hfrag1, T_SEQ);

    // ---- attention K/V precompute (full in-proj; K @ +128, V @ +256) ----
    // QKV aliases pCnnA's region; pCnnA had its last read in the first GEMM.
    gemm(Hfrag1, pAttnIn, attn_in_b, QKV, (int)MB_ROWS, 384, H);

    // ---- decoder ----
    {
        const int n = 5 * BATCH * H;
        zero_f32<<<(n + 255) / 256, 256, 0, stream>>>(state, n);
    }
    for (int step = 0; step < DSTEP; ++step) {
        dec_attention<<<BATCH, 256, 0, stream>>>(out_prev, QKV, attn_in_w, attn_in_b,
                                                 attn_out_w, attn_out_b, xcat, T_SEQ);
        dec_lstm_step<<<1, 512, 0, stream>>>(xcat, pDecWih0, pDecWhh0, pDecWih1, pDecWhh1,
                                             dec_b0, dec_b1, lin_w, lin_b,
                                             h0, c0, h1, c1, out_prev,
                                             out + (size_t)step * BATCH * VOCAB);
    }
}
